// TransformerEncoderBlock_65180423684504
// MI455X (gfx1250) — matrix operand order repaired and checked by reading, not compiled, not run on hardware
//
#include <hip/hip_runtime.h>
#include <hip/hip_bf16.h>
#include <math.h>

typedef __attribute__((ext_vector_type(16))) __bf16 v16bf;
typedef __attribute__((ext_vector_type(8)))  float  v8f;

#define SEQ_B   2
#define SEQ_N   2048
#define DMODEL  1024
#define NHEADS  16
#define DHEAD   64
#define HID     4096
#define MROWS   (SEQ_B * SEQ_N)

__device__ __forceinline__ unsigned short f2bf(float f) {
    union { float f; unsigned int u; } c; c.f = f;
    unsigned int u = c.u;
    unsigned int r = (u + 0x7FFFu + ((u >> 16) & 1u)) >> 16;  // round-to-nearest-even
    return (unsigned short)r;
}

union FragBF {
    unsigned int   u[8];
    unsigned short s[16];
    v16bf          v;
};

// ---------------------------------------------------------------------------
// fp32 -> bf16 weight conversion WITH transpose: in[K][N] -> out[N][K].
// One-time cost; eliminates all 16-bit LDS scatter from the GEMM hot loop.
// LDS-tiled 32x32 so both global read and write are coalesced.
// ---------------------------------------------------------------------------
__global__ __launch_bounds__(256)
void cvt_transpose_bf16(const float* __restrict__ in,
                        unsigned short* __restrict__ out, int K, int N) {
    __shared__ unsigned short tile[32][33];       // +1 pad: no bank conflicts
    int n0 = blockIdx.x * 32, k0 = blockIdx.y * 32;
    int tx = threadIdx.x, ty = threadIdx.y;       // (32, 8)
    #pragma unroll
    for (int t = 0; t < 4; ++t) {
        int k = ty + t * 8;
        tile[k][tx] = f2bf(in[(size_t)(k0 + k) * N + n0 + tx]);
    }
    __syncthreads();
    #pragma unroll
    for (int t = 0; t < 4; ++t) {
        int n = ty + t * 8;
        out[(size_t)(n0 + n) * K + k0 + tx] = tile[tx][n];
    }
}

// ---------------------------------------------------------------------------
// LayerNorm (one 256-thread block per row of 1024), writes bf16
// ---------------------------------------------------------------------------
__global__ __launch_bounds__(256)
void ln_bf16_kernel(const float* __restrict__ x, const float* __restrict__ g,
                    const float* __restrict__ b, unsigned short* __restrict__ out) {
    __shared__ float sm[256], sv[256];
    int row = blockIdx.x, tid = threadIdx.x;
    const float* xr = x + (size_t)row * DMODEL;
    float4 v = reinterpret_cast<const float4*>(xr)[tid];
    sm[tid] = v.x + v.y + v.z + v.w;
    sv[tid] = v.x*v.x + v.y*v.y + v.z*v.z + v.w*v.w;
    __syncthreads();
    for (int o = 128; o > 0; o >>= 1) {
        if (tid < o) { sm[tid] += sm[tid + o]; sv[tid] += sv[tid + o]; }
        __syncthreads();
    }
    float mean = sm[0] * (1.0f / DMODEL);
    float var  = sv[0] * (1.0f / DMODEL) - mean * mean;
    float rstd = rsqrtf(var + 1e-5f);
    int c0 = tid * 4;
    unsigned short* orow = out + (size_t)row * DMODEL;
    float vals[4] = {v.x, v.y, v.z, v.w};
    #pragma unroll
    for (int j = 0; j < 4; ++j)
        orow[c0 + j] = f2bf((vals[j] - mean) * rstd * g[c0 + j] + b[c0 + j]);
}

// ---------------------------------------------------------------------------
// RoPE (half = 512 across full model dim, as in reference) + bf16 convert.
// Q,K written (B,N,D) row-major; V written transposed per (b,h): [dh][seq]
// so the attention kernel's PV B-fragments gather contiguously from LDS.
// ---------------------------------------------------------------------------
__global__ __launch_bounds__(256)
void rope_cvt_kernel(const float* __restrict__ q, const float* __restrict__ k,
                     const float* __restrict__ v, const int* __restrict__ pos,
                     unsigned short* __restrict__ qb, unsigned short* __restrict__ kb,
                     unsigned short* __restrict__ vbT) {
    int row = blockIdx.x;                 // 0..MROWS-1
    int b   = row >> 11;                  // row / SEQ_N
    int n   = row & (SEQ_N - 1);
    float p = (float)pos[n];
    size_t base = (size_t)row * DMODEL;
    #pragma unroll
    for (int t = 0; t < 2; ++t) {
        int j = threadIdx.x + t * 256;    // 0..511
        float invf = __powf(10000.0f, -(float)j * (1.0f / 512.0f));
        float sn, cs; __sincosf(p * invf, &sn, &cs);
        float q1 = q[base + j], q2 = q[base + j + 512];
        qb[base + j]       = f2bf(q1 * cs - q2 * sn);
        qb[base + j + 512] = f2bf(q1 * sn + q2 * cs);
        float k1 = k[base + j], k2 = k[base + j + 512];
        kb[base + j]       = f2bf(k1 * cs - k2 * sn);
        kb[base + j + 512] = f2bf(k1 * sn + k2 * cs);
        // V: transpose to [b][h][dh][seq]
        #pragma unroll
        for (int g2 = 0; g2 < 2; ++g2) {
            int d = j + g2 * 512;
            int h = d >> 6, dh = d & 63;
            vbT[(((size_t)b * NHEADS + h) * DHEAD + dh) * SEQ_N + n] =
                f2bf(v[base + d]);
        }
    }
}

// ---------------------------------------------------------------------------
// bf16 WMMA GEMM: C[M,N] = A[M,K] @ B^T[N,K] + bias (+gelu) (+residual)
// B is pre-transposed (N-major), so BOTH tiles stage with coalesced
// global_load_b128 -> ds_store_b128 (no 16-bit scatter).
// 128x128 block tile, 256 threads = 8 waves (4x2 wave grid); each wave does
// 32x64 = 2 A-frags x 4 B-frags -> 8 wmmas per 12 ds_load_b128.
// ---------------------------------------------------------------------------
template <int ACT, bool RES, bool OUTF, bool OUTB>
__global__ __launch_bounds__(256)
void gemm_bf16_kernel(const unsigned short* __restrict__ A,
                      const unsigned short* __restrict__ BwT,
                      const float* __restrict__ bias,
                      const float* __restrict__ residual,
                      float* __restrict__ outF,
                      unsigned short* __restrict__ outB,
                      int Mn, int Nn, int Kn) {
    __shared__ __align__(16) unsigned short At[128 * 32];
    __shared__ __align__(16) unsigned short Bt[128 * 32];   // [n][k]
    int tid = threadIdx.x;
    int lane = tid & 31, w = tid >> 5;
    int lanelo = lane & 15, halfsel = lane >> 4;
    int wm = w >> 1, wn = w & 1;          // 4x2 wave grid: 32-row x 64-col per wave
    int rowBase = blockIdx.y * 128, colBase = blockIdx.x * 128;

    v8f acc[2][4];
    #pragma unroll
    for (int mt = 0; mt < 2; ++mt)
        #pragma unroll
        for (int nt = 0; nt < 4; ++nt) acc[mt][nt] = {};

    int sr = tid >> 2, ss = tid & 3;      // staging: rows sr, sr+64; 16B segment

    for (int k0 = 0; k0 < Kn; k0 += 32) {
        #pragma unroll
        for (int t = 0; t < 2; ++t) {
            int row = sr + t * 64;
            uint4 av = *reinterpret_cast<const uint4*>(
                A + (size_t)(rowBase + row) * Kn + k0 + ss * 8);
            *reinterpret_cast<uint4*>(&At[row * 32 + ss * 8]) = av;
            uint4 bv = *reinterpret_cast<const uint4*>(
                BwT + (size_t)(colBase + row) * Kn + k0 + ss * 8);
            *reinterpret_cast<uint4*>(&Bt[row * 32 + ss * 8]) = bv;
        }
        if (k0 + 32 < Kn) {
            __builtin_prefetch(A + (size_t)(rowBase + sr) * Kn + k0 + 32 + ss * 8, 0, 1);
            __builtin_prefetch(BwT + (size_t)(colBase + sr) * Kn + k0 + 32 + ss * 8, 0, 1);
        }
        __syncthreads();

        // A fragments: rows m = wm*32 + mt*16 + (lane&15); 16-bit A layout:
        // VGPR j<4: K = half*8 + 2j(+1); VGPR j>=4: K = 16 + half*8 + 2(j-4)(+1)
        FragBF af[2];
        #pragma unroll
        for (int mt = 0; mt < 2; ++mt) {
            int mA = wm * 32 + mt * 16 + lanelo;
            #pragma unroll
            for (int j = 0; j < 8; ++j) {
                int kb = ((j & 4) ? 16 : 0) + halfsel * 8 + 2 * (j & 3);
                af[mt].u[j] = *reinterpret_cast<const unsigned int*>(&At[mA * 32 + kb]);
            }
        }
        // B fragments: col n = wn*64 + nt*16 + (lane&15); K in same order as A
        FragBF bfr[4];
        #pragma unroll
        for (int nt = 0; nt < 4; ++nt) {
            int nB = wn * 64 + nt * 16 + lanelo;
            #pragma unroll
            for (int j = 0; j < 8; ++j) {
                int kb = ((j & 4) ? 16 : 0) + halfsel * 8 + 2 * (j & 3);
                bfr[nt].u[j] = *reinterpret_cast<const unsigned int*>(&Bt[nB * 32 + kb]);
            }
        }
        #pragma unroll
        for (int mt = 0; mt < 2; ++mt)
            #pragma unroll
            for (int nt = 0; nt < 4; ++nt)
                acc[mt][nt] = __builtin_amdgcn_wmma_f32_16x16x32_bf16(
                    false, af[mt].v, false, bfr[nt].v, (short)0, acc[mt][nt],
                    false, false);
        __syncthreads();
    }

    // epilogue: D layout — lane n = lane&15, VGPR e -> row half*8+e
    #pragma unroll
    for (int mt = 0; mt < 2; ++mt) {
        #pragma unroll
        for (int nt = 0; nt < 4; ++nt) {
            int col = colBase + wn * 64 + nt * 16 + lanelo;
            float bcol = bias[col];
            #pragma unroll
            for (int e = 0; e < 8; ++e) {
                int row = rowBase + wm * 32 + mt * 16 + halfsel * 8 + e;
                float vv = acc[mt][nt][e] + bcol;
                if (ACT == 1) vv = 0.5f * vv * (1.0f + erff(vv * 0.70710678f));
                size_t off = (size_t)row * Nn + col;
                if (RES)  vv += residual[off];
                if (OUTF) outF[off] = vv;
                if (OUTB) outB[off] = f2bf(vv);
            }
        }
    }
}

// ---------------------------------------------------------------------------
// Flash attention: one wave per (b, h, 16-query tile). Online softmax.
// Per 32-key step: 4 wmmas for S = Q K^T, 4 wmmas for O += P V.
// V arrives pre-transposed [dh][seq] so all fragment gathers are contiguous.
// ---------------------------------------------------------------------------
__global__ __launch_bounds__(32)
void attn_kernel(const unsigned short* __restrict__ qb,
                 const unsigned short* __restrict__ kb,
                 const unsigned short* __restrict__ vbT,
                 unsigned short* __restrict__ ctxb) {
    __shared__ __align__(16) unsigned short Qt[16 * 64];
    __shared__ __align__(16) unsigned short Kt[32 * 64];
    __shared__ __align__(16) unsigned short Vt[64 * 32];   // [dh][kv]
    __shared__ __align__(16) unsigned short Pt[16 * 32];

    int lane = threadIdx.x;
    int lanelo = lane & 15, halfsel = lane >> 4;
    int qtile = blockIdx.x;                  // 0..SEQ_N/16-1
    int bh = blockIdx.y;                     // 0..B*NHEADS-1
    int b = bh / NHEADS, h = bh % NHEADS;
    int qbase = qtile * 16;
    int hbase = h * DHEAD;
    const unsigned short* vsrc = vbT + ((size_t)bh * DHEAD) * SEQ_N;

    // stage Q tile (16 x 64 bf16)
    {
        int row = lanelo, h32 = halfsel * 32;
        const unsigned short* src =
            qb + ((size_t)(b * SEQ_N + qbase + row)) * DMODEL + hbase + h32;
        #pragma unroll
        for (int j = 0; j < 4; ++j)
            *reinterpret_cast<uint4*>(&Qt[row * 64 + h32 + j * 8]) =
                *reinterpret_cast<const uint4*>(src + j * 8);
    }
    __syncthreads();
    FragBF aq0, aq1;     // Q A-fragments for dh 0..31 and 32..63
    #pragma unroll
    for (int j = 0; j < 8; ++j) {
        int kbx = ((j & 4) ? 16 : 0) + halfsel * 8 + 2 * (j & 3);
        aq0.u[j] = *reinterpret_cast<const unsigned int*>(&Qt[lanelo * 64 + kbx]);
        aq1.u[j] = *reinterpret_cast<const unsigned int*>(&Qt[lanelo * 64 + 32 + kbx]);
    }

    v8f O0 = {}, O1 = {}, O2 = {}, O3 = {};
    float Mrow[8], Lrow[8];
    #pragma unroll
    for (int r = 0; r < 8; ++r) { Mrow[r] = -__builtin_inff(); Lrow[r] = 0.0f; }

    for (int kv0 = 0; kv0 < SEQ_N; kv0 += 32) {
        // stage K tile (32 kv x 64 dh) and V tile transposed (64 dh x 32 kv)
        {
            const unsigned short* ks =
                kb + ((size_t)(b * SEQ_N + kv0 + lane)) * DMODEL + hbase;
            #pragma unroll
            for (int j = 0; j < 8; ++j)
                *reinterpret_cast<uint4*>(&Kt[lane * 64 + j * 8]) =
                    *reinterpret_cast<const uint4*>(ks + j * 8);
            #pragma unroll
            for (int t = 0; t < 2; ++t) {
                int dh = lane + t * 32;
                #pragma unroll
                for (int j = 0; j < 4; ++j)
                    *reinterpret_cast<uint4*>(&Vt[dh * 32 + j * 8]) =
                        *reinterpret_cast<const uint4*>(
                            vsrc + (size_t)dh * SEQ_N + kv0 + j * 8);
            }
        }
        __syncthreads();

        // S = Q K^T : two 16x16 n-tiles, each accumulated over dh in 2 steps
        v8f s0 = {}, s1 = {};
        #pragma unroll
        for (int nh = 0; nh < 2; ++nh) {
            FragBF bk0, bk1;
            int n = nh * 16 + lanelo;        // key index within tile
            #pragma unroll
            for (int j = 0; j < 8; ++j) {
                int kx = ((j & 4) ? 16 : 0) + halfsel * 8 + 2 * (j & 3);   // dh index
                bk0.u[j] = *reinterpret_cast<const unsigned int*>(&Kt[n * 64 + kx]);
                bk1.u[j] = *reinterpret_cast<const unsigned int*>(&Kt[n * 64 + 32 + kx]);
            }
            v8f s = {};
            s = __builtin_amdgcn_wmma_f32_16x16x32_bf16(false, aq0.v, false, bk0.v,
                                                        (short)0, s, false, false);
            s = __builtin_amdgcn_wmma_f32_16x16x32_bf16(false, aq1.v, false, bk1.v,
                                                        (short)0, s, false, false);
            if (nh == 0) s0 = s; else s1 = s;
        }

        // online softmax (rows half*8+r replicated across each 16-lane group)
        float p0[8], p1[8], alpha[8];
        #pragma unroll
        for (int r = 0; r < 8; ++r) {
            float a = s0[r] * 0.125f, c = s1[r] * 0.125f;   // 1/sqrt(64)
            float t = fmaxf(a, c);
            #pragma unroll
            for (int off = 1; off < 16; off <<= 1)
                t = fmaxf(t, __shfl_xor(t, off, 32));
            float nm = fmaxf(Mrow[r], t);
            alpha[r] = __expf(Mrow[r] - nm);
            p0[r] = __expf(a - nm);
            p1[r] = __expf(c - nm);
            float rs = p0[r] + p1[r];
            #pragma unroll
            for (int off = 1; off < 16; off <<= 1)
                rs += __shfl_xor(rs, off, 32);
            Lrow[r] = Lrow[r] * alpha[r] + rs;
            Mrow[r] = nm;
        }
        #pragma unroll
        for (int r = 0; r < 8; ++r) {
            O0[r] *= alpha[r]; O1[r] *= alpha[r];
            O2[r] *= alpha[r]; O3[r] *= alpha[r];
            Pt[(halfsel * 8 + r) * 32 + lanelo]      = f2bf(p0[r]);
            Pt[(halfsel * 8 + r) * 32 + 16 + lanelo] = f2bf(p1[r]);
        }
        __syncthreads();

        // P as A-fragment (16x32), V as B-fragments (32x16 per dh chunk)
        FragBF ap;
        #pragma unroll
        for (int j = 0; j < 8; ++j) {
            int kbx = ((j & 4) ? 16 : 0) + halfsel * 8 + 2 * (j & 3);
            ap.u[j] = *reinterpret_cast<const unsigned int*>(&Pt[lanelo * 32 + kbx]);
        }
        #pragma unroll
        for (int c = 0; c < 4; ++c) {
            FragBF bvf;
            int dhl = c * 16 + lanelo;
            #pragma unroll
            for (int j = 0; j < 8; ++j) {
                int kx = ((j & 4) ? 16 : 0) + halfsel * 8 + 2 * (j & 3);   // kv index
                bvf.u[j] = *reinterpret_cast<const unsigned int*>(&Vt[dhl * 32 + kx]);
            }
            v8f o = (c == 0) ? O0 : (c == 1) ? O1 : (c == 2) ? O2 : O3;
            o = __builtin_amdgcn_wmma_f32_16x16x32_bf16(false, ap.v, false, bvf.v,
                                                        (short)0, o, false, false);
            if (c == 0) O0 = o; else if (c == 1) O1 = o;
            else if (c == 2) O2 = o; else O3 = o;
        }
        __syncthreads();
    }

    #pragma unroll
    for (int r = 0; r < 8; ++r) {
        float inv = 1.0f / Lrow[r];
        O0[r] *= inv; O1[r] *= inv; O2[r] *= inv; O3[r] *= inv;
    }
    #pragma unroll
    for (int c = 0; c < 4; ++c) {
        v8f o = (c == 0) ? O0 : (c == 1) ? O1 : (c == 2) ? O2 : O3;
        #pragma unroll
        for (int r = 0; r < 8; ++r) {
            int row = qbase + halfsel * 8 + r;
            ctxb[((size_t)(b * SEQ_N + row)) * DMODEL + hbase + c * 16 + lanelo] =
                f2bf(o[r]);
        }
    }
}

// ---------------------------------------------------------------------------
// Host launcher
// ---------------------------------------------------------------------------
extern "C" void kernel_launch(void* const* d_in, const int* in_sizes, int n_in,
                              void* d_out, int out_size, void* d_ws, size_t ws_size,
                              hipStream_t stream) {
    (void)in_sizes; (void)n_in; (void)out_size; (void)ws_size;
    const float* x   = (const float*)d_in[0];
    const int*   pos = (const int*)d_in[1];
    const float* Wq  = (const float*)d_in[2];
    const float* bq  = (const float*)d_in[3];
    const float* Wk  = (const float*)d_in[4];
    const float* bk  = (const float*)d_in[5];
    const float* Wv  = (const float*)d_in[6];
    const float* bv  = (const float*)d_in[7];
    const float* Wo  = (const float*)d_in[8];
    const float* bo  = (const float*)d_in[9];
    const float* g1  = (const float*)d_in[10];
    const float* b1  = (const float*)d_in[11];
    const float* g2  = (const float*)d_in[12];
    const float* b2  = (const float*)d_in[13];
    const float* W1  = (const float*)d_in[14];
    const float* bm1 = (const float*)d_in[15];
    const float* W2  = (const float*)d_in[16];
    const float* bm2 = (const float*)d_in[17];
    float* out = (float*)d_out;

    char* ws = (char*)d_ws;
    size_t off = 0;
    auto alloc = [&](size_t bytes) -> void* {
        void* p = ws + off;
        off += (bytes + 255) & ~(size_t)255;
        return p;
    };
    unsigned short* WqT  = (unsigned short*)alloc((size_t)DMODEL * DMODEL * 2);
    unsigned short* WkT  = (unsigned short*)alloc((size_t)DMODEL * DMODEL * 2);
    unsigned short* WvT  = (unsigned short*)alloc((size_t)DMODEL * DMODEL * 2);
    unsigned short* WoT  = (unsigned short*)alloc((size_t)DMODEL * DMODEL * 2);
    unsigned short* W1T  = (unsigned short*)alloc((size_t)DMODEL * HID * 2);
    unsigned short* W2T  = (unsigned short*)alloc((size_t)HID * DMODEL * 2);
    unsigned short* hb   = (unsigned short*)alloc((size_t)MROWS * DMODEL * 2);
    float*          qf   = (float*)alloc((size_t)MROWS * DMODEL * 4);
    float*          kf   = (float*)alloc((size_t)MROWS * DMODEL * 4);
    float*          vf   = (float*)alloc((size_t)MROWS * DMODEL * 4);
    unsigned short* qbb  = (unsigned short*)alloc((size_t)MROWS * DMODEL * 2);
    unsigned short* kbb  = (unsigned short*)alloc((size_t)MROWS * DMODEL * 2);
    unsigned short* vbT  = (unsigned short*)alloc((size_t)MROWS * DMODEL * 2);
    unsigned short* ctxb = (unsigned short*)alloc((size_t)MROWS * DMODEL * 2);
    float*          x2   = (float*)alloc((size_t)MROWS * DMODEL * 4);
    unsigned short* h2b  = (unsigned short*)alloc((size_t)MROWS * DMODEL * 2);
    unsigned short* hidb = (unsigned short*)alloc((size_t)MROWS * HID * 2);

    // 1) weights -> bf16, transposed to N-major
    dim3 tb(32, 8);
    cvt_transpose_bf16<<<dim3(DMODEL/32, DMODEL/32), tb, 0, stream>>>(Wq, WqT, DMODEL, DMODEL);
    cvt_transpose_bf16<<<dim3(DMODEL/32, DMODEL/32), tb, 0, stream>>>(Wk, WkT, DMODEL, DMODEL);
    cvt_transpose_bf16<<<dim3(DMODEL/32, DMODEL/32), tb, 0, stream>>>(Wv, WvT, DMODEL, DMODEL);
    cvt_transpose_bf16<<<dim3(DMODEL/32, DMODEL/32), tb, 0, stream>>>(Wo, WoT, DMODEL, DMODEL);
    cvt_transpose_bf16<<<dim3(HID/32,    DMODEL/32), tb, 0, stream>>>(W1, W1T, DMODEL, HID);
    cvt_transpose_bf16<<<dim3(DMODEL/32, HID/32),    tb, 0, stream>>>(W2, W2T, HID, DMODEL);

    // 2) LN1
    ln_bf16_kernel<<<MROWS, 256, 0, stream>>>(x, g1, b1, hb);

    // 3) Q/K/V projections (128x128 tiles)
    dim3 gqkv(DMODEL / 128, MROWS / 128);
    gemm_bf16_kernel<0, false, true, false><<<gqkv, 256, 0, stream>>>(
        hb, WqT, bq, nullptr, qf, nullptr, MROWS, DMODEL, DMODEL);
    gemm_bf16_kernel<0, false, true, false><<<gqkv, 256, 0, stream>>>(
        hb, WkT, bk, nullptr, kf, nullptr, MROWS, DMODEL, DMODEL);
    gemm_bf16_kernel<0, false, true, false><<<gqkv, 256, 0, stream>>>(
        hb, WvT, bv, nullptr, vf, nullptr, MROWS, DMODEL, DMODEL);

    // 4) RoPE + bf16 (V transposed per head)
    rope_cvt_kernel<<<MROWS, 256, 0, stream>>>(qf, kf, vf, pos, qbb, kbb, vbT);

    // 5) flash attention
    dim3 gattn(SEQ_N / 16, SEQ_B * NHEADS);
    attn_kernel<<<gattn, 32, 0, stream>>>(qbb, kbb, vbT, ctxb);

    // 6) output projection + residual(x) -> x2
    gemm_bf16_kernel<0, true, true, false><<<gqkv, 256, 0, stream>>>(
        ctxb, WoT, bo, x, x2, nullptr, MROWS, DMODEL, DMODEL);

    // 7) LN2
    ln_bf16_kernel<<<MROWS, 256, 0, stream>>>(x2, g2, b2, h2b);

    // 8) MLP1 (+gelu) -> bf16 hidden
    dim3 gm1(HID / 128, MROWS / 128);
    gemm_bf16_kernel<1, false, false, true><<<gm1, 256, 0, stream>>>(
        h2b, W1T, bm1, nullptr, nullptr, hidb, MROWS, HID, DMODEL);

    // 9) MLP2 + residual(x2) -> out
    dim3 gm2(DMODEL / 128, MROWS / 128);
    gemm_bf16_kernel<0, true, true, false><<<gm2, 256, 0, stream>>>(
        hidb, W2T, bm2, x2, out, nullptr, MROWS, DMODEL, HID);
}
